// DisenGCNZinc_78245714198785
// MI455X (gfx1250) — matrix operand order repaired and checked by reading, not compiled, not run on hardware
//
#include <hip/hip_runtime.h>
#include <math.h>

typedef __attribute__((ext_vector_type(16))) _Float16 v16h;
typedef __attribute__((ext_vector_type(8)))  float    v8f;

#define NN     20000
#define MM     320000
#define GG     512
#define DD     128
#define NATOM  28
#define NLAYER 4
#define ROUTIT 3
#define BN_EPS 1e-5f

__device__ __forceinline__ float lrelu_f(float x) { return x > 0.f ? x : 0.01f * x; }

// ---------------------------------------------------------------------------
// K0: T[32,128] = lrelu(embed(28x128, zero-padded) @ pca_w(128x128) + pca_b)
// One workgroup, 16 waves, each wave owns one 16x16 tile; K-loop of 4 WMMAs.
// Row predicate handled by clamp+mask so loads stay vectorized (b128).
// ---------------------------------------------------------------------------
__global__ void table_gemm_kernel(const float* __restrict__ E,
                                  const float* __restrict__ W,
                                  const float* __restrict__ bias,
                                  float* __restrict__ T) {
  const int wave = threadIdx.x >> 5;
  const int lane = threadIdx.x & 31;
  const int mt = wave >> 3;        // 2 row tiles
  const int nt = wave & 7;         // 8 col tiles
  const int mloc = lane & 15;
  const int hi   = lane >> 4;

  const int   m     = mt * 16 + mloc;
  const int   n     = nt * 16 + mloc;
  const float amask = (m < NATOM) ? 1.f : 0.f;     // zero-pad rows 28..31
  const int   mrow  = (m < NATOM) ? m : (NATOM - 1);

  v8f acc = {};
  for (int kk = 0; kk < 4; ++kk) {
    v16h a, b;
    // A fragment: 16x32 f16. lanes 0-15: K = [ka..ka+8) and [ka+16..ka+24)
    const int ka = kk * 32 + hi * 8;
#pragma unroll
    for (int i = 0; i < 8; ++i) {
      a[i]     = (_Float16)(amask * E[mrow * DD + ka + i]);
      a[8 + i] = (_Float16)(amask * E[mrow * DD + ka + 16 + i]);
    }
    // B fragment: 32x16 f16. lanes 0-15: K = [kb..kb+8) and [kb+16..kb+24)
    const int kb = kk * 32 + hi * 8;
#pragma unroll
    for (int i = 0; i < 8; ++i) {
      b[i]     = (_Float16)W[(kb + i) * DD + n];
      b[8 + i] = (_Float16)W[(kb + 16 + i) * DD + n];
    }
    acc = __builtin_amdgcn_wmma_f32_16x16x32_f16(false, a, false, b,
                                                 (short)0, acc, false, false);
  }
#pragma unroll
  for (int r = 0; r < 8; ++r) {
    const int row = mt * 16 + r + hi * 8;
    T[row * DD + n] = lrelu_f(acc[r] + bias[n]);
  }
}

// ---------------------------------------------------------------------------
// H[n,:] = T[atom[n],:]   (wave per node, float4 per lane)
// ---------------------------------------------------------------------------
__global__ void gather_kernel(const int* __restrict__ atoms,
                              const float4* __restrict__ T,
                              float4* __restrict__ H) {
  const int tid = blockIdx.x * blockDim.x + threadIdx.x;
  const int node = tid >> 5;
  const int lane = tid & 31;
  if (node >= NN) return;
  const int a = atoms[__builtin_amdgcn_readfirstlane(node)];
  H[node * 32 + lane] = T[a * 32 + lane];
}

// ---------------------------------------------------------------------------
// Capsule-group sum-of-squares reduce + scale (shared by the l2norm kernels)
// GSZ = lanes per capsule (dd/4 = 32/k).
// ---------------------------------------------------------------------------
template <int GSZ>
__device__ __forceinline__ float4 capsule_normalize(float4 v) {
  float ss = v.x * v.x + v.y * v.y + v.z * v.z + v.w * v.w;
#pragma unroll
  for (int off = 1; off < GSZ; off <<= 1) ss += __shfl_xor(ss, off);
  const float inv = 1.f / fmaxf(sqrtf(ss), 1e-12f);
  v.x *= inv; v.y *= inv; v.z *= inv; v.w *= inv;
  return v;
}

// x = l2norm(h) per capsule
template <int GSZ>
__global__ void l2norm_kernel(const float4* __restrict__ in,
                              float4* __restrict__ out) {
  const int tid = blockIdx.x * blockDim.x + threadIdx.x;
  const int node = tid >> 5;
  const int lane = tid & 31;
  if (node >= NN) return;
  out[node * 32 + lane] = capsule_normalize<GSZ>(in[node * 32 + lane]);
}

// u' = l2norm(segsum + x): U0 holds segment_sum only (zero-seeded)
template <int GSZ>
__global__ void l2norm_add_kernel(const float4* __restrict__ U0,
                                  const float4* __restrict__ X,
                                  float4* __restrict__ out) {
  const int tid = blockIdx.x * blockDim.x + threadIdx.x;
  const int node = tid >> 5;
  const int lane = tid & 31;
  if (node >= NN) return;
  const float4 u = U0[node * 32 + lane];
  const float4 x = X[node * 32 + lane];
  float4 v = make_float4(u.x + x.x, u.y + x.y, u.z + x.z, u.w + x.w);
  out[node * 32 + lane] = capsule_normalize<GSZ>(v);
}

// Final iteration of a layer, fused: h = lrelu(bn(l2norm(segsum + x) * snorm))
template <int GSZ>
__global__ void l2norm_post_kernel(const float4* __restrict__ U0,
                                   const float4* __restrict__ X,
                                   const float* __restrict__ snorm,
                                   const float4* __restrict__ gamma,
                                   const float4* __restrict__ beta,
                                   const float4* __restrict__ mean,
                                   const float4* __restrict__ var,
                                   float4* __restrict__ H) {
  const int tid = blockIdx.x * blockDim.x + threadIdx.x;
  const int node = tid >> 5;
  const int lane = tid & 31;
  if (node >= NN) return;
  const float4 u = U0[node * 32 + lane];
  const float4 x = X[node * 32 + lane];
  float4 v = make_float4(u.x + x.x, u.y + x.y, u.z + x.z, u.w + x.w);
  v = capsule_normalize<GSZ>(v);

  const float sn = snorm[__builtin_amdgcn_readfirstlane(node)];
  const float4 g4 = gamma[lane];
  const float4 b4 = beta[lane];
  const float4 m4 = mean[lane];
  const float4 v4 = var[lane];
  v.x = lrelu_f((v.x * sn - m4.x) * (g4.x * rsqrtf(v4.x + BN_EPS)) + b4.x);
  v.y = lrelu_f((v.y * sn - m4.y) * (g4.y * rsqrtf(v4.y + BN_EPS)) + b4.y);
  v.z = lrelu_f((v.z * sn - m4.z) * (g4.z * rsqrtf(v4.z + BN_EPS)) + b4.z);
  v.w = lrelu_f((v.w * sn - m4.w) * (g4.w * rsqrtf(v4.w + BN_EPS)) + b4.w);
  H[node * 32 + lane] = v;
}

// ---------------------------------------------------------------------------
// One routing iteration. Wave per edge; lane owns float4 of the 128-d row.
//   dot_c  = <z_c, u[trg]_c>      (capsule-group shfl-xor reduce)
//   p      = softmax_c(dot)       (cross-capsule shfl-xor max/sum, tau=1)
//   Uout[trg] += z * p            (native f32 atomics; zero-seeded Uout)
// ---------------------------------------------------------------------------
template <int GSZ>
__global__ void rout_edge_kernel(const float4* __restrict__ X,
                                 const float4* __restrict__ Uin,
                                 float* __restrict__ Uout,
                                 const int* __restrict__ src,
                                 const int* __restrict__ trg) {
  const int edge = blockIdx.x * 8 + (threadIdx.x >> 5);
  const int lane = threadIdx.x & 31;
  if (edge >= MM) return;
  const int es = __builtin_amdgcn_readfirstlane(edge);   // wave-uniform
  const int s = src[es];
  const int t = trg[es];

  const float4 z  = X[s * 32 + lane];
  const float4 ut = Uin[t * 32 + lane];

  float dot = z.x * ut.x + z.y * ut.y + z.z * ut.z + z.w * ut.w;
  // reduce within capsule group -> dot replicated across the group
#pragma unroll
  for (int off = 1; off < GSZ; off <<= 1) dot += __shfl_xor(dot, off);
  // softmax across the k capsules (butterfly over the capsule-index bits)
  float mx = dot;
#pragma unroll
  for (int off = GSZ; off < 32; off <<= 1) mx = fmaxf(mx, __shfl_xor(mx, off));
  const float e = __expf(dot - mx);
  float ssum = e;
#pragma unroll
  for (int off = GSZ; off < 32; off <<= 1) ssum += __shfl_xor(ssum, off);
  const float p = e / ssum;   // tau == 1.0

  float* dst = Uout + (size_t)t * DD + lane * 4;
  unsafeAtomicAdd(dst + 0, z.x * p);
  unsafeAtomicAdd(dst + 1, z.y * p);
  unsafeAtomicAdd(dst + 2, z.z * p);
  unsafeAtomicAdd(dst + 3, z.w * p);
}

// ---------------------------------------------------------------------------
// Per-graph sum + count (atomics into L2-resident 256 KB accumulator)
// ---------------------------------------------------------------------------
__global__ void accum_kernel(const float4* __restrict__ H,
                             const int* __restrict__ gid,
                             float* __restrict__ GS,
                             float* __restrict__ GC) {
  const int tid = blockIdx.x * blockDim.x + threadIdx.x;
  const int node = tid >> 5;
  const int lane = tid & 31;
  if (node >= NN) return;
  const int g = gid[__builtin_amdgcn_readfirstlane(node)];
  const float4 v = H[node * 32 + lane];
  float* dst = GS + (size_t)g * DD + lane * 4;
  unsafeAtomicAdd(dst + 0, v.x);
  unsafeAtomicAdd(dst + 1, v.y);
  unsafeAtomicAdd(dst + 2, v.z);
  unsafeAtomicAdd(dst + 3, v.w);
  if (lane == 0) unsafeAtomicAdd(GC + g, 1.0f);
}

__global__ void mean_kernel(const float* __restrict__ GS,
                            const float* __restrict__ GC,
                            float* __restrict__ GM) {
  const int idx = blockIdx.x * blockDim.x + threadIdx.x;
  if (idx >= GG * DD) return;
  const int g = idx >> 7;
  GM[idx] = lrelu_f(GS[idx] / fmaxf(GC[g], 1.0f));
}

// ---------------------------------------------------------------------------
// Y1[512,64] = lrelu(GM[512,128] @ reg1_w[128,64] + reg1_b)  — WMMA f16
// grid.x = 32 row tiles, 4 waves/block = 4 col tiles
// ---------------------------------------------------------------------------
__global__ void reg1_wmma_kernel(const float* __restrict__ GM,
                                 const float* __restrict__ W,
                                 const float* __restrict__ bias,
                                 float* __restrict__ Y1) {
  const int nt = threadIdx.x >> 5;
  const int lane = threadIdx.x & 31;
  const int mt = blockIdx.x;
  const int mloc = lane & 15;
  const int hi   = lane >> 4;

  v8f acc = {};
  const int m = mt * 16 + mloc;
  const int n = nt * 16 + mloc;
  for (int kk = 0; kk < 4; ++kk) {
    v16h a, b;
    const int ka = kk * 32 + hi * 8;
#pragma unroll
    for (int i = 0; i < 8; ++i) {
      a[i]     = (_Float16)GM[m * DD + ka + i];
      a[8 + i] = (_Float16)GM[m * DD + ka + 16 + i];
    }
    const int kb = kk * 32 + hi * 8;
#pragma unroll
    for (int i = 0; i < 8; ++i) {
      b[i]     = (_Float16)W[(kb + i) * 64 + n];
      b[8 + i] = (_Float16)W[(kb + 16 + i) * 64 + n];
    }
    acc = __builtin_amdgcn_wmma_f32_16x16x32_f16(false, a, false, b,
                                                 (short)0, acc, false, false);
  }
#pragma unroll
  for (int r = 0; r < 8; ++r) {
    const int row = mt * 16 + r + hi * 8;
    Y1[row * 64 + n] = lrelu_f(acc[r] + bias[n]);
  }
}

// ---------------------------------------------------------------------------
// out[g] = Y1[g,:] . reg2_w + reg2_b   (wave per graph)
// ---------------------------------------------------------------------------
__global__ void reg2_kernel(const float* __restrict__ Y1,
                            const float* __restrict__ W,
                            const float* __restrict__ bias,
                            float* __restrict__ out) {
  const int tid = blockIdx.x * blockDim.x + threadIdx.x;
  const int g = tid >> 5;
  const int lane = tid & 31;
  if (g >= GG) return;
  const int e = lane * 2;
  float p = Y1[g * 64 + e] * W[e] + Y1[g * 64 + e + 1] * W[e + 1];
#pragma unroll
  for (int off = 1; off < 32; off <<= 1) p += __shfl_xor(p, off);
  if (lane == 0) out[g] = p + bias[0];
}

// ---------------------------------------------------------------------------
extern "C" void kernel_launch(void* const* d_in, const int* in_sizes, int n_in,
                              void* d_out, int out_size, void* d_ws, size_t ws_size,
                              hipStream_t stream) {
  const int*   x_atoms  = (const int*)  d_in[0];
  const int*   src      = (const int*)  d_in[1];
  const int*   trg      = (const int*)  d_in[2];
  const float* snorm    = (const float*)d_in[3];
  const int*   gid      = (const int*)  d_in[4];
  const float* embed    = (const float*)d_in[5];
  const float* pca_w    = (const float*)d_in[6];
  const float* pca_b    = (const float*)d_in[7];
  const float* bn_gamma = (const float*)d_in[8];
  const float* bn_beta  = (const float*)d_in[9];
  const float* bn_mean  = (const float*)d_in[10];
  const float* bn_var   = (const float*)d_in[11];
  const float* reg1_w   = (const float*)d_in[12];
  const float* reg1_b   = (const float*)d_in[13];
  const float* reg2_w   = (const float*)d_in[14];
  const float* reg2_b   = (const float*)d_in[15];
  float* out = (float*)d_out;

  // workspace carve-up (256B aligned)
  char* ws = (char*)d_ws;
  size_t off = 0;
  auto carve = [&](size_t bytes) -> void* {
    void* p = ws + off;
    off = (off + bytes + 255) & ~(size_t)255;
    return p;
  };
  const size_t nodeBytes = (size_t)NN * DD * sizeof(float);
  float* H  = (float*)carve(nodeBytes);
  float* X  = (float*)carve(nodeBytes);
  float* U0 = (float*)carve(nodeBytes);   // segment-sum accumulator (zeroed)
  float* U1 = (float*)carve(nodeBytes);   // normalized u (iters 1..ROUTIT-1)
  float* T  = (float*)carve(32 * DD * sizeof(float));
  float* GS = (float*)carve((size_t)GG * DD * sizeof(float));
  float* GC = (float*)carve(GG * sizeof(float));
  float* GM = (float*)carve((size_t)GG * DD * sizeof(float));
  float* Y1 = (float*)carve((size_t)GG * 64 * sizeof(float));
  (void)ws_size; (void)in_sizes; (void)n_in; (void)out_size;

  const dim3 blk256(256);
  const dim3 nodeWaveGrid((NN * 32 + 255) / 256);      // wave per node
  const dim3 edgeGrid((MM + 7) / 8);                   // wave per edge, 8/block

  // (a) collapsed embed@pca table + gather
  table_gemm_kernel<<<1, 512, 0, stream>>>(embed, pca_w, pca_b, T);
  gather_kernel<<<nodeWaveGrid, blk256, 0, stream>>>(x_atoms, (const float4*)T,
                                                     (float4*)H);

  // (b) 4 disentangle layers, 3 routing iterations each
  for (int layer = 0; layer < NLAYER; ++layer) {
    const bool k8 = (layer < 2);   // k=8 -> GSZ=4 ; k=4 -> GSZ=8

    if (k8)
      l2norm_kernel<4><<<nodeWaveGrid, blk256, 0, stream>>>((const float4*)H,
                                                            (float4*)X);
    else
      l2norm_kernel<8><<<nodeWaveGrid, blk256, 0, stream>>>((const float4*)H,
                                                            (float4*)X);

    const float* u_in = X;
    for (int it = 0; it < ROUTIT; ++it) {
      hipMemsetAsync(U0, 0, nodeBytes, stream);        // segsum accumulator
      const bool last = (it == ROUTIT - 1);
      if (k8) {
        rout_edge_kernel<4><<<edgeGrid, blk256, 0, stream>>>(
            (const float4*)X, (const float4*)u_in, U0, src, trg);
        if (!last)
          l2norm_add_kernel<4><<<nodeWaveGrid, blk256, 0, stream>>>(
              (const float4*)U0, (const float4*)X, (float4*)U1);
        else
          l2norm_post_kernel<4><<<nodeWaveGrid, blk256, 0, stream>>>(
              (const float4*)U0, (const float4*)X, snorm,
              (const float4*)(bn_gamma + layer * DD),
              (const float4*)(bn_beta + layer * DD),
              (const float4*)(bn_mean + layer * DD),
              (const float4*)(bn_var + layer * DD), (float4*)H);
      } else {
        rout_edge_kernel<8><<<edgeGrid, blk256, 0, stream>>>(
            (const float4*)X, (const float4*)u_in, U0, src, trg);
        if (!last)
          l2norm_add_kernel<8><<<nodeWaveGrid, blk256, 0, stream>>>(
              (const float4*)U0, (const float4*)X, (float4*)U1);
        else
          l2norm_post_kernel<8><<<nodeWaveGrid, blk256, 0, stream>>>(
              (const float4*)U0, (const float4*)X, snorm,
              (const float4*)(bn_gamma + layer * DD),
              (const float4*)(bn_beta + layer * DD),
              (const float4*)(bn_mean + layer * DD),
              (const float4*)(bn_var + layer * DD), (float4*)H);
      }
      u_in = U1;
    }
  }

  // (c) per-graph mean + MLP head
  hipMemsetAsync(GS, 0, (size_t)GG * DD * sizeof(float), stream);
  hipMemsetAsync(GC, 0, GG * sizeof(float), stream);
  accum_kernel<<<nodeWaveGrid, blk256, 0, stream>>>((const float4*)H, gid, GS, GC);
  mean_kernel<<<(GG * DD + 255) / 256, blk256, 0, stream>>>(GS, GC, GM);
  reg1_wmma_kernel<<<32, 128, 0, stream>>>(GM, reg1_w, reg1_b, Y1);
  reg2_kernel<<<(GG * 32 + 255) / 256, blk256, 0, stream>>>(Y1, reg2_w, reg2_b, out);
}
